// ScaleDotProductAttention_13091060318507
// MI455X (gfx1250) — compile-verified, matrix operand order checked by reading, not run on hardware
//
#include <hip/hip_runtime.h>
#include <hip/hip_bf16.h>

typedef __attribute__((ext_vector_type(16))) _Float16 v16h;
typedef __attribute__((ext_vector_type(8)))  float    v8f;

namespace {
constexpr int   Lq     = 2048;
constexpr int   Dd     = 64;
constexpr int   NT     = Lq / 16;      // 128 key tiles of 16
constexpr float kScale = 0.125f;       // 1/sqrt(64)
constexpr int   WPB    = 4;            // waves per block
}

// K-dim offset of packed element pair p (0..7) for the 16-bit 16x32 WMMA
// fragment layout: VGPR p holds K = {base, base+1}, lanes 16-31 add +8.
__device__ __forceinline__ int kbase_pair(int p, int g) {
  return ((p & 3) << 1) + ((p >> 2) << 4) + (g << 3);
}

// Load a 16x32 f16 fragment (A layout; also B layout for K^T) from a
// row-major f32 matrix. Element e of a lane = src[row*ld + koff + kpat(e,g)].
// Addresses of each pair are 8-byte aligned -> float2 loads.
__device__ __forceinline__ v16h load_frag_k32(const float* __restrict__ src,
                                              int row, int ld, int koff, int g) {
  v16h r;
#pragma unroll
  for (int p = 0; p < 8; ++p) {
    const float2 f = *(const float2*)(src + (size_t)row * ld + koff + kbase_pair(p, g));
    r[2 * p]     = (_Float16)f.x;
    r[2 * p + 1] = (_Float16)f.y;
  }
  return r;
}

__global__ __launch_bounds__(WPB * 32)
void sdpa_fwd_kernel(const float* __restrict__ q,
                     const float* __restrict__ k,
                     const float* __restrict__ v,
                     float* __restrict__ ctx,
                     float* __restrict__ att) {
  // Per-wave f32 staging of a 16x32 probability tile (2 KB/wave).
  // Used both for the C-fragment -> A-fragment transpose feeding P*V and
  // to emit the attention output as fully coalesced 128B row segments.
  __shared__ float ldsP[WPB][16][32];

  const int lane = threadIdx.x & 31;
  const int w    = threadIdx.x >> 5;
  const int gw   = blockIdx.x * WPB + w;   // 0..4095 : (b*h, q-tile)
  const int qt   = gw & (NT - 1);          // q tile within (b,h)
  const int bh   = gw >> 7;                // 0..31

  const int g = lane >> 4;                 // half-wave group
  const int n = lane & 15;                 // column / row-in-tile index

  const float* qb = q + (size_t)bh * Lq * Dd;
  const float* kb = k + (size_t)bh * Lq * Dd;
  const float* vb = v + (size_t)bh * Lq * Dd;
  float* attb = att + (size_t)bh * Lq * Lq + (size_t)(qt * 16) * Lq;
  float* ctxb = ctx + (size_t)bh * Lq * Dd + (size_t)(qt * 16) * Dd;

  // Q fragments (A layout), K-dim split 0..31 / 32..63
  const v16h aq0 = load_frag_k32(qb, qt * 16 + n, Dd, 0,  g);
  const v16h aq1 = load_frag_k32(qb, qt * 16 + n, Dd, 32, g);

  // ---------------- Pass 1: online softmax stats ----------------
  // C fragment: lane (g,n) VGPR r holds score(row = r+8g, col = tile*16+n).
  float m8[8], l8[8];
#pragma unroll
  for (int r = 0; r < 8; ++r) { m8[r] = -3.0e38f; l8[r] = 0.0f; }

  for (int t = 0; t < NT; ++t) {
    const v16h bk0 = load_frag_k32(kb, t * 16 + n, Dd, 0,  g);
    const v16h bk1 = load_frag_k32(kb, t * 16 + n, Dd, 32, g);
    v8f c = {};
    c = __builtin_amdgcn_wmma_f32_16x16x32_f16(false, aq0, false, bk0, (short)0, c, false, false);
    c = __builtin_amdgcn_wmma_f32_16x16x32_f16(false, aq1, false, bk1, (short)0, c, false, false);
#pragma unroll
    for (int r = 0; r < 8; ++r) {
      const float s  = c[r] * kScale;
      const float nm = fmaxf(m8[r], s);
      l8[r] = l8[r] * __expf(m8[r] - nm) + __expf(s - nm);
      m8[r] = nm;
    }
  }
  // Combine (m, l) pairs across the 16 lanes holding each row (xor 1,2,4,8
  // never crosses the half-wave boundary).
#pragma unroll
  for (int r = 0; r < 8; ++r) {
#pragma unroll
    for (int mask = 1; mask <= 8; mask <<= 1) {
      const float om = __shfl_xor(m8[r], mask, 32);
      const float ol = __shfl_xor(l8[r], mask, 32);
      const float nm = fmaxf(m8[r], om);
      l8[r] = l8[r] * __expf(m8[r] - nm) + ol * __expf(om - nm);
      m8[r] = nm;
    }
  }
  float inv8[8];
#pragma unroll
  for (int r = 0; r < 8; ++r) inv8[r] = 1.0f / l8[r];

  // -------- Pass 2: recompute scores, emit probs, accumulate P*V --------
  v8f acc[4] = {v8f{}, v8f{}, v8f{}, v8f{}};   // context cols 0-15,16-31,32-47,48-63

  for (int ch = 0; ch < NT / 2; ++ch) {        // 32 keys per chunk
#pragma unroll
    for (int tc = 0; tc < 2; ++tc) {
      const int t = ch * 2 + tc;
      const v16h bk0 = load_frag_k32(kb, t * 16 + n, Dd, 0,  g);
      const v16h bk1 = load_frag_k32(kb, t * 16 + n, Dd, 32, g);
      v8f c = {};
      c = __builtin_amdgcn_wmma_f32_16x16x32_f16(false, aq0, false, bk0, (short)0, c, false, false);
      c = __builtin_amdgcn_wmma_f32_16x16x32_f16(false, aq1, false, bk1, (short)0, c, false, false);
#pragma unroll
      for (int r = 0; r < 8; ++r) {
        const float pr = __expf(c[r] * kScale - m8[r]) * inv8[r];
        ldsP[w][r + 8 * g][tc * 16 + n] = pr;              // stage f32 prob tile
      }
    }
    __builtin_amdgcn_wave_barrier();   // wave-local LDS RAW; DS in-order + dscnt

    // Attention output: 16 rows x 32 cols f32, emitted as float4 per lane so
    // each 8-lane group covers one full 128B-aligned row segment.
#pragma unroll
    for (int i = 0; i < 4; ++i) {
      const int row = i * 4 + (lane >> 3);
      const int c4  = (lane & 7) * 4;
      const float4 pv4 = *(const float4*)&ldsP[w][row][c4];
      *(float4*)&attb[(size_t)row * Lq + ch * 32 + c4] = pv4;
    }

    // A fragment of the 16x32 probability tile from LDS (f32 -> f16)
    v16h pa;
#pragma unroll
    for (int p = 0; p < 8; ++p) {
      const float2 f = *(const float2*)&ldsP[w][n][kbase_pair(p, g)];
      pa[2 * p]     = (_Float16)f.x;
      pa[2 * p + 1] = (_Float16)f.y;
    }
    // V B-fragments: K-dim along V rows, 4 groups of 16 output columns
#pragma unroll
    for (int cg = 0; cg < 4; ++cg) {
      v16h bv;
#pragma unroll
      for (int p = 0; p < 8; ++p) {
        const int kr = kbase_pair(p, g);
        bv[2 * p]     = (_Float16)vb[(size_t)(ch * 32 + kr)     * Dd + cg * 16 + n];
        bv[2 * p + 1] = (_Float16)vb[(size_t)(ch * 32 + kr + 1) * Dd + cg * 16 + n];
      }
      acc[cg] = __builtin_amdgcn_wmma_f32_16x16x32_f16(false, pa, false, bv, (short)0, acc[cg], false, false);
    }
  }

  // Context write (probs already normalized before PV)
#pragma unroll
  for (int cg = 0; cg < 4; ++cg)
#pragma unroll
    for (int r = 0; r < 8; ++r)
      ctxb[(size_t)(r + 8 * g) * Dd + cg * 16 + n] = acc[cg][r];
}

extern "C" void kernel_launch(void* const* d_in, const int* in_sizes, int n_in,
                              void* d_out, int out_size, void* d_ws, size_t ws_size,
                              hipStream_t stream) {
  (void)in_sizes; (void)n_in; (void)out_size; (void)d_ws; (void)ws_size;
  const float* q = (const float*)d_in[0];
  const float* k = (const float*)d_in[1];
  const float* v = (const float*)d_in[2];

  constexpr int B = 2, H = 16;
  float* ctx = (float*)d_out;                                  // [B,H,L,D]
  float* att = (float*)d_out + (size_t)B * H * Lq * Dd;        // [B,H,L,L]

  const int totalWaves = B * H * NT;            // 4096 q-tiles
  const int blocks     = totalWaves / WPB;      // 1024
  sdpa_fwd_kernel<<<blocks, WPB * 32, 0, stream>>>(q, k, v, ctx, att);
}
